// Decoder_MLP_32873679683849
// MI455X (gfx1250) — matrix operand order repaired and checked by reading, not compiled, not run on hardware
//
#include <hip/hip_runtime.h>
#include <stdint.h>

typedef __attribute__((ext_vector_type(16))) __bf16 v16bf;
typedef __attribute__((ext_vector_type(8)))  float  v8f;
typedef unsigned short u16;
typedef unsigned int   u32;

// ---------- bf16 helpers (round-to-nearest-even) ----------
__device__ __forceinline__ u16 f2bf(float f) {
  u32 u = __float_as_uint(f);
  u32 r = (u + 0x7FFFu + ((u >> 16) & 1u)) >> 16;
  return (u16)r;
}
__device__ __forceinline__ float bf2f(u16 h) {
  return __uint_as_float(((u32)h) << 16);
}

// =====================================================================
// Pack x (25088 x 512 f32, row-major) into WMMA A-fragment tiles (bf16).
// A tile = 16x32; per-lane chunk = 16 contiguous bf16 (32B) at
// tile*512 + lane*16.  Lane = (m%16) + 16*hs;  hs selects K-halves.
// Thread = (r, kt, hs): 25088*16*2 = 802816 threads.
// =====================================================================
__global__ void pack_x_kernel(const float* __restrict__ X, u16* __restrict__ A) {
  int tid = blockIdx.x * blockDim.x + threadIdx.x;
  int hs = tid & 1;
  int kt = (tid >> 1) & 15;
  int r  = tid >> 5;                       // 0..25087
  int run0 = kt * 32 + hs * 8;             // K runs: [run0..run0+7], [run0+16..run0+23]
  const float* xp = X + (size_t)r * 512;
  u16* op = A + (((size_t)(r >> 4) * 16 + kt) << 9) + (((r & 15) + 16 * hs) << 4);
#pragma unroll
  for (int j = 0; j < 8; ++j) {
    op[j]     = f2bf(xp[run0 + j]);
    op[8 + j] = f2bf(xp[run0 + 16 + j]);
  }
}

// =====================================================================
// Pack weight W (Nrows x K f32, row-major) into B-fragment tiles (bf16).
// B tile = 32x16 (KxN); lane = (n%16) + 16*kh; lane chunk = 16 K values
// (k = kt*32 + kh*8 + 0..7, then kt*32 + 16 + kh*8 + 0..7) of column n.
// Tiles stored (nt, kt) with kt inner.  Threads = Nrows*KT*2.
// =====================================================================
__global__ void pack_w_kernel(const float* __restrict__ W, u16* __restrict__ B,
                              int K, int KT) {
  int tid = blockIdx.x * blockDim.x + threadIdx.x;
  int kh   = tid & 1;
  int rest = tid >> 1;
  int kt   = rest % KT;
  int n    = rest / KT;
  const float* wp = W + (size_t)n * K + kt * 32 + kh * 8;
  u16* op = B + (((size_t)(n >> 4) * KT + kt) << 9) + (((n & 15) + 16 * kh) << 4);
#pragma unroll
  for (int j = 0; j < 16; ++j) op[j] = f2bf(wp[j < 8 ? j : j + 8]);
}

// =====================================================================
// bf16 WMMA GEMM:  C(bf16, row-major, M x Nd) = A(packed) * B(packed)
// Wave tile = 32x64 (2 M-tiles x 4 N-tiles, 8 f32 accumulators).
// 8 waves / block -> block covers 256 rows (16 M-tiles) x 64 cols.
// B operand staged block-wide into LDS in 64KB chunks (4 nt x 16 kt)
// via async global->LDS copies (ASYNCcnt), shared by all 8 waves.
// A streamed per-wave from L2 with prefetch.
// =====================================================================
__global__ void __launch_bounds__(256)
wmma_gemm_kernel(const u16* __restrict__ A, const u16* __restrict__ B,
                 u16* __restrict__ C, int KT, int Nd) {
  __shared__ __align__(64) unsigned char ldsB[65536];
  int tid  = threadIdx.x;
  int lane = tid & 31;
  int wave = tid >> 5;
  int mt0  = blockIdx.x * 16 + wave * 2;   // first of 2 M-tiles for this wave
  int ntb  = blockIdx.y * 4;               // first of 4 N-tiles for this block

  const u16* Ap0 = A + (((size_t)mt0 * KT) << 9) + (lane << 4);
  const u16* Ap1 = Ap0 + ((size_t)KT << 9);

  const v8f vzero = {0.f, 0.f, 0.f, 0.f, 0.f, 0.f, 0.f, 0.f};
  v8f acc[8];
#pragma unroll
  for (int i = 0; i < 8; ++i) acc[i] = vzero;

  for (int kc = 0; kc < KT; kc += 16) {
    // ---- cooperative async stage of B[ntb..ntb+3][kc..kc+15] (64KB) ----
    // 4096 16-byte units; each of the 256 threads moves 16 units.
#pragma unroll
    for (int j = 0; j < 16; ++j) {
      int u    = tid + 256 * j;            // 0..4095
      int i_nt = u >> 10;                  // which N-tile column
      int rem  = u & 1023;                 // 16B unit within 16KB run
      const u16* src = B + (((size_t)(ntb + i_nt) * KT + kc) << 9) + rem * 8;
      u32 ldst = (u32)(size_t)(void*)&ldsB[(size_t)u << 4];
      asm volatile("global_load_async_to_lds_b128 %0, %1, off"
                   :: "v"(ldst), "v"(src) : "memory");
    }
    asm volatile("s_wait_asynccnt 0" ::: "memory");
    __syncthreads();

    // ---- 16 k-steps from LDS-resident B ----
    for (int ktc = 0; ktc < 16; ++ktc) {
      size_t ko = (size_t)(kc + ktc) << 9;
      v16bf a0 = *(const v16bf*)(Ap0 + ko);
      v16bf a1 = *(const v16bf*)(Ap1 + ko);
      __builtin_prefetch(Ap0 + ko + 512, 0, 1);   // global_prefetch next A tiles
      __builtin_prefetch(Ap1 + ko + 512, 0, 1);
#pragma unroll
      for (int i = 0; i < 4; ++i) {
        v16bf b = *(const v16bf*)(ldsB + (((i << 4) + ktc) << 10) + (lane << 5));
        acc[i]     = __builtin_amdgcn_wmma_f32_16x16x32_bf16(false, a0, false, b, (short)0, acc[i],     false, false);
        acc[4 + i] = __builtin_amdgcn_wmma_f32_16x16x32_bf16(false, a1, false, b, (short)0, acc[4 + i], false, false);
      }
    }
    __syncthreads();   // protect LDS before next chunk overwrites it
  }

  // C/D layout: VGPR v, lanes 0-15 -> M=v, lanes 16-31 -> M=v+8; N = lane%16
  int cbase = ntb * 16 + (lane & 15);
  int rsel  = (lane >> 4) << 3;
#pragma unroll
  for (int m = 0; m < 2; ++m) {
    int rbase = (mt0 + m) * 16 + rsel;
#pragma unroll
    for (int v = 0; v < 8; ++v) {
      size_t ro = (size_t)(rbase + v) * Nd + cbase;
      C[ro]      = f2bf(acc[4 * m + 0][v]);
      C[ro + 16] = f2bf(acc[4 * m + 1][v]);
      C[ro + 32] = f2bf(acc[4 * m + 2][v]);
      C[ro + 48] = f2bf(acc[4 * m + 3][v]);
    }
  }
}

// =====================================================================
// Column stats, stage 1: per (256-row block, channel) partial sum/sumsq.
// grid = (Ch/256, R/256).  Deterministic (no atomics).
// =====================================================================
__global__ void col_stats_partial_kernel(const u16* __restrict__ X,
                                         float* __restrict__ part, int Ch) {
  int c = blockIdx.x * 256 + threadIdx.x;
  size_t base = (size_t)blockIdx.y * 256 * Ch + c;
  float s = 0.f, q = 0.f;
  for (int i = 0; i < 256; ++i) {
    float x = bf2f(X[base + (size_t)i * Ch]);
    s += x; q += x * x;
  }
  size_t o = ((size_t)blockIdx.y * Ch + c) * 2;
  part[o] = s; part[o + 1] = q;
}

// stage 2: fold partials -> per-channel affine (a = g*rstd, b = beta - mean*a)
__global__ void col_stats_final_kernel(const float* __restrict__ part, int RB,
                                       float invR, int Ch,
                                       const float* __restrict__ gamma,
                                       const float* __restrict__ beta,
                                       float* __restrict__ ab) {
  int c = blockIdx.x * 256 + threadIdx.x;
  if (c >= Ch) return;
  float s = 0.f, q = 0.f;
  for (int i = 0; i < RB; ++i) {
    s += part[((size_t)i * Ch + c) * 2];
    q += part[((size_t)i * Ch + c) * 2 + 1];
  }
  float mean = s * invR;
  float var  = q * invR - mean * mean;
  float rstd = rsqrtf(var + 1e-5f);
  float a = gamma[c] * rstd;
  ab[2 * c]     = a;
  ab[2 * c + 1] = beta[c] - mean * a;
}

// =====================================================================
// Fused BN1 + 4-step LIF + reshape-quirk permute + A-fragment repack.
// Dest chunk = (row r0' in [0,6272), kt in [0,64), hs in {0,1}) of the
// GEMM2 A matrix (25088 x 2048).  The reference's reshape(-1,H,N) quirk:
// dest feature h_ at row n_ comes from source flat = h_*196 + n_, which
// is exactly the row-major (n*2048 + h) offset inside the tb-slice of H.
// LIF couples t at fixed (b, flat); BN channel = flat % 2048.
// Threads: r0' fastest -> coalesced gathers from H.
// =====================================================================
__global__ void bn_lif_pack_kernel(const u16* __restrict__ H,
                                   const float* __restrict__ ab,
                                   u16* __restrict__ A2) {
  int tid = blockIdx.x * blockDim.x + threadIdx.x;   // < 802816 = 6272*128
  int r0    = tid % 6272;                            // b*196 + n_
  int chunk = tid / 6272;                            // 0..127
  int hs = chunk & 1;
  int kt = chunk >> 1;                               // 0..63
  int b  = r0 / 196;
  int n_ = r0 % 196;
  int lane = (r0 & 15) + 16 * hs;

  float v[16], aa[16], bb[16];
  int flat[16];
#pragma unroll
  for (int j = 0; j < 16; ++j) {
    int kk = (j < 8 ? j : j + 8) + hs * 8;           // A-fragment K map
    int h_ = kt * 32 + kk;
    int f  = h_ * 196 + n_;                          // source offset in slice
    flat[j] = f;
    int ch = f & 2047;                               // original BN channel
    aa[j] = ab[2 * ch];
    bb[j] = ab[2 * ch + 1];
    v[j]  = 0.f;
  }
#pragma unroll
  for (int t = 0; t < 4; ++t) {
    size_t slice = (size_t)(t * 32 + b) * 401408;    // tb * N*H
    u32 packed[8];
#pragma unroll
    for (int j = 0; j < 16; ++j) {
      float x  = bf2f(H[slice + flat[j]]);
      float y  = aa[j] * x + bb[j];
      float vv = 0.5f * (v[j] + y);                  // v += (y - v)/tau, tau=2
      bool sp  = vv >= 1.0f;                         // hard threshold v_th=1
      u32 s = sp ? 0x3F80u : 0u;                     // bf16 1.0 / 0.0
      v[j] = sp ? 0.f : vv;                          // hard reset to 0
      if (j & 1) packed[j >> 1] |= s << 16; else packed[j >> 1] = s;
    }
    size_t rp = (size_t)(r0 + t * 6272);             // dest row
    u32* op = (u32*)(A2 + (((rp >> 4) * 64 + kt) << 9) + ((size_t)lane << 4));
#pragma unroll
    for (int j = 0; j < 8; ++j) op[j] = packed[j];
  }
}

// =====================================================================
// Fused BN2 + LIF + final reshape-quirk -> fp32 spikes to d_out.
// out[tb, n_, c_] = spike at source flat = c_*196 + n_ (of the (n,c) plane).
// =====================================================================
__global__ void bn_lif_out_kernel(const u16* __restrict__ O,
                                  const float* __restrict__ ab,
                                  float* __restrict__ out) {
  int tid = blockIdx.x * blockDim.x + threadIdx.x;   // < 3211264
  int c_ = tid & 511;
  int r0 = tid >> 9;                                 // b*196 + n_
  int b  = r0 / 196;
  int n_ = r0 % 196;
  int flat = c_ * 196 + n_;
  int ch   = flat & 511;
  float a = ab[2 * ch], bb = ab[2 * ch + 1], v = 0.f;
#pragma unroll
  for (int t = 0; t < 4; ++t) {
    size_t slice = (size_t)(t * 32 + b) * 100352;    // tb * N*C
    float x  = bf2f(O[slice + flat]);
    float y  = a * x + bb;
    float vv = 0.5f * (v + y);
    bool sp  = vv >= 1.0f;
    out[(size_t)(r0 + t * 6272) * 512 + c_] = sp ? 1.0f : 0.0f;
    v = sp ? 0.f : vv;
  }
}

// =====================================================================
extern "C" void kernel_launch(void* const* d_in, const int* in_sizes, int n_in,
                              void* d_out, int out_size, void* d_ws, size_t ws_size,
                              hipStream_t stream) {
  const float* x  = (const float*)d_in[0];
  const float* w1 = (const float*)d_in[1];
  const float* g1 = (const float*)d_in[2];
  const float* b1 = (const float*)d_in[3];
  const float* w2 = (const float*)d_in[4];
  const float* g2 = (const float*)d_in[5];
  const float* b2 = (const float*)d_in[6];
  (void)in_sizes; (void)n_in; (void)out_size;

  // workspace layout (bytes)
  const size_t oA1 = 0;                  // 25088*512*2   = 25,690,112
  const size_t oB1 = 25690112;           // 2048*512*2    =  2,097,152
  const size_t oH  = 27787264;           // 25088*2048*2  = 102,760,448
  const size_t oPT = 130547712;          // 98*2048*2*4   =  1,605,632
  const size_t oAB = 132153344;          // 2048*2*4      =     16,384
  const size_t oA2 = 132169728;          // 25088*2048*2  = 102,760,448
  const size_t oB2 = 234930176;          // 2048*512*2    =  2,097,152
  const size_t oO  = 237027328;          // 25088*512*2   = 25,690,112
  const size_t need = 262717440;
  if (ws_size < need) return;

  char* ws = (char*)d_ws;
  u16*   A1   = (u16*)(ws + oA1);
  u16*   B1   = (u16*)(ws + oB1);
  u16*   Hm   = (u16*)(ws + oH);
  float* part = (float*)(ws + oPT);
  float* ab   = (float*)(ws + oAB);
  u16*   A2   = (u16*)(ws + oA2);
  u16*   B2   = (u16*)(ws + oB2);
  u16*   Om   = (u16*)(ws + oO);

  // 1) pack operands to bf16 WMMA fragment layouts
  pack_x_kernel<<<3136, 256, 0, stream>>>(x, A1);                 // 802816 thr
  pack_w_kernel<<<256, 256, 0, stream>>>(w1, B1, 512, 16);        // H=2048 rows
  pack_w_kernel<<<256, 256, 0, stream>>>(w2, B2, 2048, 64);       // C=512 rows

  // 2) GEMM1: (25088x512)*(512x2048) -> Hm bf16
  wmma_gemm_kernel<<<dim3(98, 32), 256, 0, stream>>>(A1, B1, Hm, 16, 2048);

  // 3) BN1 stats (deterministic two-stage), then fused BN+LIF+permute+pack
  col_stats_partial_kernel<<<dim3(8, 98), 256, 0, stream>>>(Hm, part, 2048);
  col_stats_final_kernel<<<8, 256, 0, stream>>>(part, 98, 1.0f / 25088.0f, 2048, g1, b1, ab);
  bn_lif_pack_kernel<<<3136, 256, 0, stream>>>(Hm, ab, A2);

  // 4) GEMM2: (25088x2048)*(2048x512) -> Om bf16
  wmma_gemm_kernel<<<dim3(98, 8), 256, 0, stream>>>(A2, B2, Om, 64, 512);

  // 5) BN2 stats + fused BN+LIF -> fp32 spikes (with final reshape quirk)
  col_stats_partial_kernel<<<dim3(2, 98), 256, 0, stream>>>(Om, part, 512);
  col_stats_final_kernel<<<2, 256, 0, stream>>>(part, 98, 1.0f / 25088.0f, 512, g2, b2, ab);
  bn_lif_out_kernel<<<12544, 256, 0, stream>>>(Om, ab, (float*)d_out);
}
